// SpatialContextSet_3444563771604
// MI455X (gfx1250) — matrix operand order repaired and checked by reading, not compiled, not run on hardware
//
#include <hip/hip_runtime.h>
#include <hip/hip_bf16.h>

// SpatialContextSet fused forward for gfx1250 (MI455X).
// bf16 WMMA (v_wmma_f32_16x16x32_bf16) for all heavy GEMMs, f32 accumulate,
// f32 VALU for LN / sigmoid / GELU. Attention-K and W2 GEMMs folded
// algebraically: heavy work = W_stat, Wv, W1 GEMMs per token.
// GEMM restructured: each wave owns one M-tile, A-fragments hoisted into
// registers and reused across all 8 N-tiles (8x fewer ds_load_b128 per wmma).

typedef __attribute__((ext_vector_type(16))) __bf16 v16bf;
typedef __attribute__((ext_vector_type(8)))  __bf16 v8bf;
typedef __attribute__((ext_vector_type(8)))  float  v8f;

#define LN_EPS 1e-5f
#define ATT_SCALE 0.125f   // DH^-0.5, DH=64

// ---- WMMA fragment helpers (CDNA5 16x16x32 bf16 layouts, wave32) ----------
// A (16x32 bf16): lane L<16 holds row M=L, K = {0..7, 16..23}; lane L+16 same
// row, K = {8..15, 24..31}.  -> two 16B LDS loads per lane.
__device__ inline v16bf load_A(const __bf16* s, int stride, int m0, int k0) {
  const int L = threadIdx.x & 31;
  const int m = L & 15, hh = L >> 4;
  const __bf16* p = s + (size_t)(m0 + m) * stride + k0 + hh * 8;
  union { v16bf v; v8bf h[2]; } u;
  u.h[0] = *(const v8bf*)p;
  u.h[1] = *(const v8bf*)(p + 16);
  return u.v;
}

// B (32x16 bf16): lane L holds column N = L&15, K = (L>>4)*8 + {0..7, 16..23}.
// Weights are stored TRANSPOSED (bf16, [N][K]) so K is contiguous ->
// two global_load_b128 per lane.
__device__ inline v16bf load_B(const __bf16* g, int stride, int n0, int koff) {
  const int L = threadIdx.x & 31;
  const int n = L & 15, kh = (L >> 4) * 8;
  const __bf16* p = g + (size_t)(n0 + n) * stride + koff + kh;
  union { v16bf v; v8bf h[2]; } u;
  u.h[0] = *(const v8bf*)p;
  u.h[1] = *(const v8bf*)(p + 16);
  return u.v;
}

__device__ inline v8f wmma_bf16(v16bf a, v16bf b, v8f c) {
  return __builtin_amdgcn_wmma_f32_16x16x32_bf16(false, a, false, b,
                                                 (short)0, c, false, false);
}

// Tiled GEMM: sOut[64][256] = sA[64][K](bf16, LDS) @ WT(bf16 global, [256][K])
// + bias.  8 waves: wave w owns M-tile (w>>1), N-tiles (w&1)*128 .. +112.
// A-fragments for the whole K loop are loaded once and kept in registers.
// C/D layout: vgpr r, lane L -> (m = r + 8*(L>>4), n = L&15).
template<int M, int N, int K>
__device__ inline void gemm_bf16(const __bf16* __restrict__ sA, int sa_stride,
                                 const __bf16* __restrict__ WT, int wt_stride,
                                 float* __restrict__ sOut, int out_stride,
                                 const float* __restrict__ bias) {
  static_assert(M == 64 && N == 256, "fixed tiling");
  const int L = threadIdx.x & 31;
  const int w = threadIdx.x >> 5;
  const int m0    = (w >> 1) * 16;
  const int nbase = (w & 1) * 128;
  constexpr int KF = K / 32;
  v16bf afrag[KF];
#pragma unroll
  for (int kk = 0; kk < KF; kk++)
    afrag[kk] = load_A(sA, sa_stride, m0, kk * 32);
  const int hh = L >> 4;
  for (int j = 0; j < 8; j++) {
    const int n0 = nbase + j * 16;
    v8f acc = {};
#pragma unroll
    for (int kk = 0; kk < KF; kk++)
      acc = wmma_bf16(afrag[kk], load_B(WT, wt_stride, n0, kk * 32), acc);
    const int n = n0 + (L & 15);
    const float bb = bias ? bias[n] : 0.f;
#pragma unroll
    for (int r = 0; r < 8; r++)
      sOut[(size_t)(m0 + r + 8 * hh) * out_stride + n] = acc[r] + bb;
  }
}

// Row LayerNorm over 64 rows x 256 cols, f32 in -> bf16 out. 4 lanes per row,
// partial sums combined with wave32 shuffles.
__device__ inline void ln_64x256(const float* sIn, int in_stride,
                                 __bf16* sOut, int out_stride,
                                 const float* g, const float* bsh) {
  const int row = threadIdx.x >> 2, part = threadIdx.x & 3;
  const float* p = sIn + (size_t)row * in_stride + part * 64;
  float s = 0.f, ss = 0.f;
  for (int i = 0; i < 64; i++) { float x = p[i]; s += x; ss += x * x; }
  s += __shfl_xor(s, 1); ss += __shfl_xor(ss, 1);
  s += __shfl_xor(s, 2); ss += __shfl_xor(ss, 2);
  const float mean = s * (1.f / 256.f);
  const float rstd = rsqrtf(ss * (1.f / 256.f) - mean * mean + LN_EPS);
  __bf16* q = sOut + (size_t)row * out_stride + part * 64;
  for (int i = 0; i < 64; i++) {
    const int c = part * 64 + i;
    q[i] = (__bf16)((p[i] - mean) * rstd * g[c] + bsh[c]);
  }
}

// ---- weight conversion pre-pass -------------------------------------------
__global__ void scs_convT(const float* __restrict__ src, __bf16* __restrict__ dst,
                          int K, int N) {          // src[K][N] -> dst[N][K]
  int i = blockIdx.x * 256 + threadIdx.x;
  if (i < K * N) { int k = i / N, n = i % N; dst[(size_t)n * K + k] = (__bf16)src[i]; }
}
__global__ void scs_conv(const float* __restrict__ src, __bf16* __restrict__ dst, int cnt) {
  int i = blockIdx.x * 256 + threadIdx.x;
  if (i < cnt) dst[i] = (__bf16)src[i];
}

// ---- Kernel A: h0, q, r, lb (64 rows of B per block) ----------------------
__global__ __launch_bounds__(256) void scs_kA(
    const float* __restrict__ h_dyn, const __bf16* __restrict__ WtD,
    const __bf16* __restrict__ WtQ,  const __bf16* __restrict__ WkR,
    const float* __restrict__ b_dyn, const float* __restrict__ g_nd,
    const float* __restrict__ b_nd,  const float* __restrict__ bq,
    const float* __restrict__ bk,
    float* __restrict__ h0_out, float* __restrict__ q_out,
    float* __restrict__ r_out,  float* __restrict__ lb_out) {
  __shared__ __align__(16) __bf16 pool[64 * 528];   // h_dyn(520) / sH(264) / sQ(264)
  __shared__ __align__(16) float  sF[64 * 256];
  const int tid = threadIdx.x;
  const int r0  = blockIdx.x * 64;

  // load h_dyn 64x512 -> bf16, stride 520
  const float4* hb = (const float4*)(h_dyn + (size_t)r0 * 512);
  for (int i = tid; i < 8192; i += 256) {
    float4 v = hb[i];
    int e = i * 4, row = e >> 9, col = e & 511;
    __bf16* d = pool + (size_t)row * 520 + col;
    d[0] = (__bf16)v.x; d[1] = (__bf16)v.y; d[2] = (__bf16)v.z; d[3] = (__bf16)v.w;
  }
  __syncthreads();
  gemm_bf16<64, 256, 512>(pool, 520, WtD, 512, sF, 256, b_dyn);   // h0
  __syncthreads();
  for (int i = tid; i < 16384; i += 256) {
    int row = i >> 8, c = i & 255;
    h0_out[(size_t)(r0 + row) * 256 + c] = sF[row * 256 + c];
  }
  __bf16* sH = pool;                       // h_dyn dead -> reuse
  ln_64x256(sF, 256, sH, 264, g_nd, b_nd);
  __syncthreads();
  gemm_bf16<64, 256, 256>(sH, 264, WtQ, 256, sF, 256, bq);        // q
  __syncthreads();
  __bf16* sQ = pool + 16896;               // bytes [33792,67584)
  for (int i = tid; i < 16384; i += 256) {
    int row = i >> 8, c = i & 255;
    float v = sF[row * 256 + c];
    q_out[(size_t)(r0 + row) * 256 + c] = v;
    sQ[(size_t)row * 264 + c] = (__bf16)v;
  }
  { // lb[b,h] = q[b,h,:] . bk_h
    int row = tid >> 2, h = tid & 3;
    float acc = 0.f;
    for (int d = 0; d < 64; d++) acc += sF[row * 256 + h * 64 + d] * bk[h * 64 + d];
    lb_out[(size_t)(r0 + row) * 4 + h] = acc;
  }
  __syncthreads();
  // r[b,h,c] = sum_d Wk[c, h*64+d] * q[b,h,d]  (B operand = row-major Wk bf16)
  const int L = tid & 31;
  const int w = tid >> 5;
  const int m0 = (w >> 1) * 16, nbase = (w & 1) * 128;
  const int hh = L >> 4;
  for (int h = 0; h < 4; h++) {
    v16bf a0 = load_A(sQ, 264, m0, h * 64);
    v16bf a1 = load_A(sQ, 264, m0, h * 64 + 32);
    for (int j = 0; j < 8; j++) {
      const int n0 = nbase + j * 16;
      v8f acc = {};
      acc = wmma_bf16(a0, load_B(WkR, 256, n0, h * 64), acc);
      acc = wmma_bf16(a1, load_B(WkR, 256, n0, h * 64 + 32), acc);
      const int n = n0 + (L & 15);
#pragma unroll
      for (int r = 0; r < 8; r++)
        r_out[(size_t)(r0 + m0 + r + 8 * hh) * 1024 + h * 256 + n] = acc[r];
    }
  }
}

// ---- Kernel B: fused token pipeline, one batch element per block ----------
__global__ __launch_bounds__(256) void scs_kB(
    const float* __restrict__ x_stat, const float* __restrict__ r_in,
    const float* __restrict__ lb_in,
    const __bf16* __restrict__ WtS, const __bf16* __restrict__ WtV,
    const __bf16* __restrict__ Wt1, const __bf16* __restrict__ Wt2,
    const float* __restrict__ b_stat, const float* __restrict__ g_ns,
    const float* __restrict__ b_ns,   const float* __restrict__ bvv,
    const float* __restrict__ g_mlp,  const float* __restrict__ b_mlp,
    const float* __restrict__ b1,     const float* __restrict__ b2,
    float* __restrict__ z_out, float* __restrict__ wmean_out) {
  __shared__ __align__(16) __bf16 sS[64 * 264];   // s_ln, later t1
  __shared__ __align__(16) __bf16 sX[64 * 72];
  __shared__ __align__(16) float  sF[64 * 256];
  __shared__ float sR[4 * 256];
  __shared__ float sW[4 * 64];
  __shared__ float sA[4 * 256];
  __shared__ float sLB[4], sWn[4];
  const int tid = threadIdx.x;
  const int b   = blockIdx.x;

  // stage 0: loads
  const float4* xb = (const float4*)(x_stat + (size_t)b * 4096);
  for (int i = tid; i < 1024; i += 256) {
    float4 v = xb[i];
    int e = i * 4, row = e >> 6, col = e & 63;
    __bf16* d = sX + (size_t)row * 72 + col;
    d[0] = (__bf16)v.x; d[1] = (__bf16)v.y; d[2] = (__bf16)v.z; d[3] = (__bf16)v.w;
  }
  for (int i = tid; i < 1024; i += 256) sR[i] = r_in[(size_t)b * 1024 + i];
  if (tid < 4) sLB[tid] = lb_in[(size_t)b * 4 + tid];
  __syncthreads();

  // s = x_stat @ W_stat + b_stat ; LN -> bf16
  gemm_bf16<64, 256, 64>(sX, 72, WtS, 64, sF, 256, b_stat);
  __syncthreads();
  ln_64x256(sF, 256, sS, 264, g_ns, b_ns);
  __syncthreads();

  // logits[h,k] = (s_ln[k] . r[h] + lb[h]) * SCALE ; sigmoid
  {
    const int h = tid >> 6, k = tid & 63;
    float acc = 0.f;
    for (int c = 0; c < 256; c++) acc += (float)sS[(size_t)k * 264 + c] * sR[h * 256 + c];
    float lg = (acc + sLB[h]) * ATT_SCALE;
    sW[h * 64 + k] = 1.f / (1.f + __expf(-lg));
  }
  __syncthreads();
  if (tid < 4) { float s = 0.f; for (int k = 0; k < 64; k++) s += sW[tid * 64 + k]; sWn[tid] = s; }
  __syncthreads();
  { const int h = tid >> 6, k = tid & 63; sW[h * 64 + k] *= 1.f / (sWn[h] + 1e-6f); }
  __syncthreads();
  if (tid < 4) sWn[tid] = sWn[tid] / (sWn[tid] + 1e-6f);   // sum of normalized w
  if (tid < 64) {
    float m = 0.25f * (sW[tid] + sW[64 + tid] + sW[128 + tid] + sW[192 + tid]);
    wmean_out[(size_t)b * 64 + tid] = m;
  }
  // v_flat = s_ln @ Wv + bv ; LN -> t (reuse sS after sync)
  gemm_bf16<64, 256, 256>(sS, 264, WtV, 256, sF, 256, bvv);
  __syncthreads();
  ln_64x256(sF, 256, sS, 264, g_mlp, b_mlp);
  __syncthreads();
  // u = t @ W1 + b1 ; GELU(exact) -> t1 (reuse sS)
  gemm_bf16<64, 256, 256>(sS, 264, Wt1, 256, sF, 256, b1);
  __syncthreads();
  for (int i = tid; i < 16384; i += 256) {
    float x = sF[i];
    sS[(size_t)(i >> 8) * 264 + (i & 255)] =
        (__bf16)(0.5f * x * (1.f + erff(x * 0.70710678f)));
  }
  __syncthreads();
  // a[h,c] = sum_k w[h,k] * t1[k,c]
  {
    const int c = tid;
    float a0 = 0.f, a1 = 0.f, a2 = 0.f, a3 = 0.f;
    for (int k = 0; k < 64; k++) {
      float t = (float)sS[(size_t)k * 264 + c];
      a0 += sW[k] * t; a1 += sW[64 + k] * t; a2 += sW[128 + k] * t; a3 += sW[192 + k] * t;
    }
    sA[c] = a0; sA[256 + c] = a1; sA[512 + c] = a2; sA[768 + c] = a3;
  }
  __syncthreads();
  // z[n] = a[h(n)] . W2t[n] + wn[h] * b2[n]
  {
    const int n = tid, h = n >> 6;
    const __bf16* w2r = Wt2 + (size_t)n * 256;
    const float* ah = sA + h * 256;
    float acc = 0.f;
    for (int c = 0; c < 256; c++) acc += ah[c] * (float)w2r[c];
    z_out[(size_t)b * 256 + n] = acc + sWn[h] * b2[n];
  }
}

// ---- Kernel C: c = h0 + res_scale * (z @ Wo + bo) -------------------------
__global__ __launch_bounds__(256) void scs_kC(
    const float* __restrict__ z_in, const __bf16* __restrict__ WtO,
    const float* __restrict__ bo,   const float* __restrict__ h0_in,
    const float* __restrict__ res_scale, float* __restrict__ c_out) {
  __shared__ __align__(16) __bf16 sZ[64 * 264];
  __shared__ __align__(16) float  sF[64 * 256];
  const int tid = threadIdx.x;
  const int r0  = blockIdx.x * 64;
  const float4* zb = (const float4*)(z_in + (size_t)r0 * 256);
  for (int i = tid; i < 4096; i += 256) {
    float4 v = zb[i];
    int e = i * 4, row = e >> 8, col = e & 255;
    __bf16* d = sZ + (size_t)row * 264 + col;
    d[0] = (__bf16)v.x; d[1] = (__bf16)v.y; d[2] = (__bf16)v.z; d[3] = (__bf16)v.w;
  }
  __syncthreads();
  gemm_bf16<64, 256, 256>(sZ, 264, WtO, 256, sF, 256, bo);
  __syncthreads();
  const float rs = *res_scale;
  for (int i = tid; i < 16384; i += 256) {
    int row = i >> 8, c = i & 255;
    size_t gi = (size_t)(r0 + row) * 256 + c;
    c_out[gi] = h0_in[gi] + rs * sF[row * 256 + c];
  }
}

extern "C" void kernel_launch(void* const* d_in, const int* in_sizes, int n_in,
                              void* d_out, int out_size, void* d_ws, size_t ws_size,
                              hipStream_t stream) {
  (void)in_sizes; (void)n_in; (void)out_size; (void)ws_size;
  const float* h_dyn   = (const float*)d_in[0];
  const float* x_stat  = (const float*)d_in[1];
  const float* W_dyn   = (const float*)d_in[2];
  const float* b_dyn   = (const float*)d_in[3];
  const float* W_stat  = (const float*)d_in[4];
  const float* b_stat  = (const float*)d_in[5];
  const float* g_ndyn  = (const float*)d_in[6];
  const float* b_ndyn  = (const float*)d_in[7];
  const float* g_nstat = (const float*)d_in[8];
  const float* b_nstat = (const float*)d_in[9];
  const float* Wq = (const float*)d_in[10]; const float* bq = (const float*)d_in[11];
  const float* Wk = (const float*)d_in[12]; const float* bk = (const float*)d_in[13];
  const float* Wv = (const float*)d_in[14]; const float* bv = (const float*)d_in[15];
  const float* g_mlp = (const float*)d_in[16]; const float* b_mlp = (const float*)d_in[17];
  const float* W1 = (const float*)d_in[18]; const float* b1 = (const float*)d_in[19];
  const float* W2 = (const float*)d_in[20]; const float* b2 = (const float*)d_in[21];
  const float* Wo = (const float*)d_in[22]; const float* bo = (const float*)d_in[23];
  const float* res_scale = (const float*)d_in[24];

  char* ws = (char*)d_ws;
  size_t off = 0;
  auto carve = [&](size_t bytes) {
    char* p = ws + off;
    off = (off + bytes + 255) & ~(size_t)255;
    return p;
  };
  __bf16* WtD = (__bf16*)carve(256 * 512 * 2);   // W_dyn^T  [256][512]
  __bf16* WtS = (__bf16*)carve(256 * 64 * 2);    // W_stat^T [256][64]
  __bf16* WtQ = (__bf16*)carve(256 * 256 * 2);   // Wq^T
  __bf16* WkR = (__bf16*)carve(256 * 256 * 2);   // Wk row-major bf16
  __bf16* WtV = (__bf16*)carve(256 * 256 * 2);   // Wv^T
  __bf16* Wt1 = (__bf16*)carve(256 * 256 * 2);   // W1^T
  __bf16* Wt2 = (__bf16*)carve(256 * 256 * 2);   // W2^T
  __bf16* WtO = (__bf16*)carve(256 * 256 * 2);   // Wo^T
  float* h0 = (float*)carve((size_t)4096 * 256 * 4);
  float* qf = (float*)carve((size_t)4096 * 256 * 4);
  float* rf = (float*)carve((size_t)4096 * 1024 * 4);
  float* lb = (float*)carve((size_t)4096 * 4 * 4);
  float* zf = (float*)carve((size_t)4096 * 256 * 4);

  scs_convT<<<512, 256, 0, stream>>>(W_dyn, WtD, 512, 256);
  scs_convT<<<64, 256, 0, stream>>>(W_stat, WtS, 64, 256);
  scs_convT<<<256, 256, 0, stream>>>(Wq, WtQ, 256, 256);
  scs_conv <<<256, 256, 0, stream>>>(Wk, WkR, 65536);
  scs_convT<<<256, 256, 0, stream>>>(Wv, WtV, 256, 256);
  scs_convT<<<256, 256, 0, stream>>>(W1, Wt1, 256, 256);
  scs_convT<<<256, 256, 0, stream>>>(W2, Wt2, 256, 256);
  scs_convT<<<256, 256, 0, stream>>>(Wo, WtO, 256, 256);

  scs_kA<<<64, 256, 0, stream>>>(h_dyn, WtD, WtQ, WkR, b_dyn, g_ndyn, b_ndyn,
                                 bq, bk, h0, qf, rf, lb);
  scs_kB<<<4096, 256, 0, stream>>>(x_stat, rf, lb, WtS, WtV, Wt1, Wt2,
                                   b_stat, g_nstat, b_nstat, bv, g_mlp, b_mlp,
                                   b1, b2, zf, (float*)d_out + (size_t)4096 * 256);
  scs_kC<<<64, 256, 0, stream>>>(zf, WtO, bo, h0, res_scale, (float*)d_out);
}
